// SelectiveLinear_781684048387
// MI455X (gfx1250) — matrix operand order repaired and checked by reading, not compiled, not run on hardware
//
#include <hip/hip_runtime.h>

// Problem constants (from reference): B=2048, IN=512, OUT=512, H=16
#define B_N   2048
#define IN_F  512
#define OUT_F 512
#define H_N   16
#define MROWS   64                 // rows (M) per block = 4 WMMA row-groups
#define MGROUPS 4
#define MT_MAX  (B_N / MROWS)      // 32 max m-tiles per head
#define NT_BLOCK 128               // 8 waves * 16 cols per block
#define TPB 256                    // 8 wave32s
#define KCHUNKS (IN_F / 32)        // 16 K-chunks of 32

typedef __attribute__((ext_vector_type(16))) __bf16          v16bf;
typedef __attribute__((ext_vector_type(16))) unsigned short  v16us;
typedef __attribute__((ext_vector_type(8)))  unsigned int    v8u;
typedef __attribute__((ext_vector_type(8)))  float           v8f;

// ---------------- Kernel 1: zero per-head counters ----------------
__global__ void sl_zero_counts(int* counts) {
    if (threadIdx.x < H_N) counts[threadIdx.x] = 0;
}

// ---- Kernel 2: bucketize rows by head; also emit index passthrough ----
__global__ void sl_bucketize(const int* __restrict__ idx,
                             int* __restrict__ counts,
                             int* __restrict__ rowlist,
                             float* __restrict__ out_tail) {
    int b = blockIdx.x * blockDim.x + threadIdx.x;
    if (b >= B_N) return;
    int h = idx[b] & (H_N - 1);
    int pos = atomicAdd(&counts[h], 1);
    rowlist[h * B_N + pos] = b;
    out_tail[b] = (float)h;                    // tuple output #2: X_head_idx
}

// ---------------- Kernel 3: grouped GEMM via bf16x3 WMMA ----------------
// 64 rows x 128 cols per block: each global B fragment feeds 12 WMMAs.
__global__ __launch_bounds__(TPB)
void sl_gemm(const float* __restrict__ X,
             const float* __restrict__ W,      // [H, IN, OUT]
             const float* __restrict__ bias,   // [H, OUT]
             const int*   __restrict__ counts,
             const int*   __restrict__ rowlist,
             float*       __restrict__ out,
             float*       __restrict__ dump) { // scratch sink for padding rows
    const int h  = blockIdx.x & (H_N - 1);
    const int mt = blockIdx.x >> 4;
    const int cnt = counts[h];
    if (mt * MROWS >= cnt) return;             // uniform early exit
    const int nb0 = blockIdx.y * NT_BLOCK;

    extern __shared__ __attribute__((aligned(32))) char smem[];
    v16us* AHi = (v16us*)smem;                           // 4*16*32 frags, 64 KB
    v16us* ALo = AHi + MGROUPS * KCHUNKS * 32;           // 64 KB
    int*   rows_s = (int*)(ALo + MGROUPS * KCHUNKS * 32);// 64 ints

    const int tid = threadIdx.x;
    if (tid < MROWS) {
        int bi = mt * MROWS + tid;
        rows_s[tid] = (bi < cnt) ? rowlist[h * B_N + bi] : -1;
    }
    __syncthreads();

    // Stage X: coalesced global reads, hi/lo bf16 split, scatter directly
    // into per-lane WMMA A-fragment slots (compute reads = ds_load_b128).
    {
        unsigned short* AHiS = (unsigned short*)AHi;
        unsigned short* ALoS = (unsigned short*)ALo;
        for (int i = tid; i < MROWS * IN_F; i += TPB) {
            int r = i >> 9;                    // row 0..63
            int k = i & (IN_F - 1);            // 0..511
            int g = rows_s[r];
            float x = (g >= 0) ? X[(size_t)g * IN_F + k] : 0.0f;
            unsigned u  = __float_as_uint(x);
            unsigned hb = u & 0xFFFF0000u;                 // truncated bf16 hi
            float    lo = x - __uint_as_float(hb);         // exact residual
            int grp = r >> 4, m = r & 15;
            int chunk = k >> 5, kk = k & 31;
            int lh = (kk >> 4) & 1;
            int e  = kk & 15;
            int slot = (((grp * KCHUNKS + chunk) << 5) + m + (lh << 4)) * 16 + e;
            AHiS[slot] = (unsigned short)(hb >> 16);
            ALoS[slot] = (unsigned short)(__float_as_uint(lo) >> 16);
        }
    }
    __syncthreads();

    const int wave     = tid >> 5;
    const int lane     = tid & 31;
    const int laneHalf = lane >> 4;
    const int lane16   = lane & 15;
    const int nb       = nb0 + wave * 16;

    v8f c0 = {}, c1 = {}, c2 = {}, c3 = {};    // 4 row-group accumulators
    const float* wcol = W + ((size_t)h * IN_F) * OUT_F + nb + lane16
                          + (size_t)(laneHalf ? 16 : 0) * OUT_F;

    // Unroll x2: second chunk's independent loads + perm/fma-mix packing
    // fills the WMMA WAR-hazard slots (4 co-exec NOPs for *_BF16) instead
    // of v_nop.
#pragma unroll 2
    for (int ch = 0; ch < KCHUNKS; ++ch) {
        // ---- B fragment (32x16 bf16): pack hi/lo pairs with v_perm_b32
        v8u bh32, bl32;
        const float* wp = wcol + (size_t)(ch << 5) * OUT_F;
#pragma unroll
        for (int e2 = 0; e2 < 8; ++e2) {
            float w0 = wp[(size_t)(2 * e2)     * OUT_F];
            float w1 = wp[(size_t)(2 * e2 + 1) * OUT_F];
            unsigned u0 = __float_as_uint(w0);
            unsigned u1 = __float_as_uint(w1);
            unsigned h0 = u0 & 0xFFFF0000u;
            unsigned h1 = u1 & 0xFFFF0000u;
            float l0 = w0 - __uint_as_float(h0);
            float l1 = w1 - __uint_as_float(h1);
            bh32[e2] = __builtin_amdgcn_perm(h1, h0, 0x07060302u);
            bl32[e2] = __builtin_amdgcn_perm(__float_as_uint(l1),
                                             __float_as_uint(l0), 0x07060302u);
        }
        v16bf bh = __builtin_bit_cast(v16bf, bh32);
        v16bf bl = __builtin_bit_cast(v16bf, bl32);

        // ---- 4 row-groups reuse this B fragment: 12 WMMAs per fetch
#define SL_GRP(G, CACC)                                                        \
        {                                                                      \
            v16bf ah = __builtin_bit_cast(v16bf, AHi[((G * KCHUNKS + ch) << 5) + lane]); \
            v16bf al = __builtin_bit_cast(v16bf, ALo[((G * KCHUNKS + ch) << 5) + lane]); \
            CACC = __builtin_amdgcn_wmma_f32_16x16x32_bf16(false, ah, false, bh, (short)0, CACC, false, false); \
            CACC = __builtin_amdgcn_wmma_f32_16x16x32_bf16(false, ah, false, bl, (short)0, CACC, false, false); \
            CACC = __builtin_amdgcn_wmma_f32_16x16x32_bf16(false, al, false, bh, (short)0, CACC, false, false); \
        }
        SL_GRP(0, c0)
        SL_GRP(1, c1)
        SL_GRP(2, c2)
        SL_GRP(3, c3)
#undef SL_GRP
    }

    // Epilogue: branch-free scatter. Padding rows store to a scratch sink.
    float bv = bias[h * OUT_F + nb + lane16];
    const size_t colOff = (size_t)nb + lane16;
#define SL_EPI(G, CACC)                                                        \
    _Pragma("unroll")                                                          \
    for (int r = 0; r < 8; ++r) {                                              \
        int M = (G << 4) + laneHalf * 8 + r;                                   \
        int g = rows_s[M];                                                     \
        float* p = (g >= 0) ? (out + (size_t)g * OUT_F + colOff)               \
                            : (dump + tid);                                    \
        *p = CACC[r] + bv;                                                     \
    }
    SL_EPI(0, c0)
    SL_EPI(1, c1)
    SL_EPI(2, c2)
    SL_EPI(3, c3)
#undef SL_EPI
}

extern "C" void kernel_launch(void* const* d_in, const int* in_sizes, int n_in,
                              void* d_out, int out_size, void* d_ws, size_t ws_size,
                              hipStream_t stream) {
    const float* X    = (const float*)d_in[0];
    const int*   idx  = (const int*)d_in[1];     // head index per row (0..15)
    const float* W    = (const float*)d_in[2];   // [H, IN, OUT] f32
    const float* bias = (const float*)d_in[3];   // [H, OUT] f32
    float* out = (float*)d_out;

    int*   counts  = (int*)d_ws;                 // H counters
    int*   rowlist = counts + 64;                // H*B row indices
    float* dump    = (float*)(rowlist + H_N * B_N); // TPB-float scratch sink

    sl_zero_counts<<<1, 64, 0, stream>>>(counts);
    sl_bucketize<<<B_N / TPB, TPB, 0, stream>>>(idx, counts, rowlist,
                                                out + (size_t)B_N * OUT_F);

    // Dynamic LDS: 2*64KB A fragments + row list (CDNA5 WGP has 320KB LDS)
    size_t smem = (size_t)MGROUPS * KCHUNKS * 32 * 32 * 2 + MROWS * sizeof(int);
    dim3 grid(H_N * MT_MAX, OUT_F / NT_BLOCK);   // (512, 4); inactive tiles exit
    sl_gemm<<<grid, TPB, smem, stream>>>(X, W, bias, counts, rowlist, out, dump);
}
